// ScaledDotProductAttention_34763465294507
// MI455X (gfx1250) — matrix operand order repaired and checked by reading, not compiled, not run on hardware
//
#include <hip/hip_runtime.h>
#include <stdint.h>

#define BB 16
#define SS 2048
#define DD 128

constexpr int QT    = 16;        // q rows per wave
constexpr int KT    = 32;        // k cols per tile
constexpr int NT    = SS / KT;   // 64 k-tiles
constexpr int WAVES = 4;         // waves per block (4 consecutive q tiles, same batch)
constexpr int TPB   = WAVES * 32;
constexpr int KP    = 136;       // K-tile LDS pitch in halves (272B = 17*16B, bank-skewed)
constexpr int VP    = 40;        // V^T-tile LDS pitch in halves (80B)
constexpr int PP    = 40;        // P scratch LDS pitch in halves (80B)

typedef _Float16 v4h  __attribute__((ext_vector_type(4)));
typedef _Float16 v8h  __attribute__((ext_vector_type(8)));
typedef _Float16 v16h __attribute__((ext_vector_type(16)));
typedef float    v8f  __attribute__((ext_vector_type(8)));
typedef int      v4i  __attribute__((ext_vector_type(4)));

typedef __attribute__((address_space(1))) v4i gv4i;   // global int4
typedef __attribute__((address_space(3))) v4i lv4i;   // LDS int4

__device__ inline v8f zero8() {
  v8f r;
  #pragma unroll
  for (int i = 0; i < 8; ++i) r[i] = 0.0f;
  return r;
}

__device__ inline v16h ld16h2(const _Float16* p0, const _Float16* p1) {
  v8h lo = *(const v8h*)p0;
  v8h hi = *(const v8h*)p1;
  return __builtin_shufflevector(lo, hi, 0,1,2,3,4,5,6,7,8,9,10,11,12,13,14,15);
}

__device__ inline float redmax16(float x) {
  #pragma unroll
  for (int m = 8; m >= 1; m >>= 1) x = fmaxf(x, __shfl_xor(x, m, 32));
  return x;   // xor masks 1..8 stay within the 16-lane half of the wave32
}

__device__ inline float redsum16(float x) {
  #pragma unroll
  for (int m = 8; m >= 1; m >>= 1) x += __shfl_xor(x, m, 32);
  return x;
}

__device__ inline v8f wmma_f16(v16h a, v16h b, v8f c) {
  return __builtin_amdgcn_wmma_f32_16x16x32_f16(false, a, false, b,
                                                (short)0, c, false, false);
}

// ---- CDNA5 async global -> LDS (ASYNCcnt-tracked, no VGPR round trip) ----
__device__ inline void async_b128(const _Float16* g, _Float16* l) {
#if __has_builtin(__builtin_amdgcn_global_load_async_to_lds_b128)
  __builtin_amdgcn_global_load_async_to_lds_b128((gv4i*)g, (lv4i*)l, 0, 0);
#else
  uint32_t lo = (uint32_t)(uintptr_t)l;   // LDS byte offset = addr[31:0]
  asm volatile("global_load_async_to_lds_b128 %0, %1, off"
               :: "v"(lo), "v"(g) : "memory");
#endif
}

#if __has_builtin(__builtin_amdgcn_s_wait_asynccnt)
#define WAIT_ASYNC(n) __builtin_amdgcn_s_wait_asynccnt(n)
#else
#define WAIT_ASYNC(n) asm volatile("s_wait_asynccnt %0" :: "i"(n) : "memory")
#endif

// ---------------- prepass: fp32 -> f16, Q pre-scaled by 1/sqrt(D) ----------------
__global__ void cvt_qk_kernel(const float4* __restrict__ q, const float4* __restrict__ k,
                              v4h* __restrict__ qh, v4h* __restrict__ kh,
                              int n4, float qscale) {
  int i = blockIdx.x * blockDim.x + threadIdx.x;
  if (i >= n4) return;
  float4 a = q[i];
  float4 b = k[i];
  v4h oa, ob;
  oa[0] = (_Float16)(a.x * qscale); oa[1] = (_Float16)(a.y * qscale);
  oa[2] = (_Float16)(a.z * qscale); oa[3] = (_Float16)(a.w * qscale);
  ob[0] = (_Float16)b.x; ob[1] = (_Float16)b.y;
  ob[2] = (_Float16)b.z; ob[3] = (_Float16)b.w;
  qh[i] = oa;
  kh[i] = ob;
}

// tiled transpose: v[b][s][d] (f32) -> vt[b][d][s] (f16)
__global__ void cvt_v_kernel(const float* __restrict__ v, _Float16* __restrict__ vt) {
  __shared__ _Float16 tile[32][33];
  const int b  = blockIdx.z;
  const int s0 = blockIdx.x * 32;
  const int d0 = blockIdx.y * 32;
  const int tx = threadIdx.x;           // 0..31
  const int ty = threadIdx.y;           // 0..7
  #pragma unroll
  for (int r = 0; r < 4; ++r)
    tile[ty + 8*r][tx] = (_Float16)v[((size_t)b*SS + s0 + ty + 8*r)*DD + d0 + tx];
  __syncthreads();
  #pragma unroll
  for (int r = 0; r < 4; ++r)
    vt[((size_t)b*DD + d0 + ty + 8*r)*SS + s0 + tx] = tile[tx][ty + 8*r];
}

// ---------------- flash attention ----------------
__global__ __launch_bounds__(TPB)
void fa_kernel(const _Float16* __restrict__ qh, const _Float16* __restrict__ kh,
               const _Float16* __restrict__ vt, const uint8_t* __restrict__ mask,
               float* __restrict__ out) {
  __shared__ __align__(16) _Float16 Ks[2][KT * KP];        // K tile   [k][d], double buffered
  __shared__ __align__(16) _Float16 Vs[2][DD * VP];        // V^T tile [d][k], double buffered
  __shared__ __align__(16) _Float16 Ps[WAVES][QT * PP];    // per-wave P scratch

  const int lane = threadIdx.x & 31;
  const int wave = threadIdx.x >> 5;
  const int tile = blockIdx.x * WAVES + wave;
  const int b    = tile / (SS / QT);
  const int q0   = (tile % (SS / QT)) * QT;
  const int n    = lane & 15;           // N index / column within fragment
  const int h    = lane >> 4;           // lane half selector
  const int t    = threadIdx.x;

  const _Float16* kbatch = kh + (size_t)b * SS * DD;
  const _Float16* vbatch = vt + (size_t)b * DD * SS;

  // async cooperative stage of one k-tile (8 x b128 per thread)
  auto stage = [&](int kt, int buf) {
    const _Float16* ksrc = kbatch + (size_t)(kt * KT) * DD;
    const _Float16* vsrc = vbatch + kt * KT;
    _Float16* kb = &Ks[buf][0];
    _Float16* vb = &Vs[buf][0];
    #pragma unroll
    for (int s = 0; s < 4; ++s) {           // K: 32 rows x 16 chunks of 16B
      int i = t + s * TPB;
      int r = i >> 4, cc = i & 15;
      async_b128(ksrc + r * DD + cc * 8, kb + r * KP + cc * 8);
    }
    #pragma unroll
    for (int s = 0; s < 4; ++s) {           // V^T: 128 rows x 4 chunks of 16B
      int i = t + s * TPB;
      int r = i >> 2, cc = i & 3;
      async_b128(vsrc + (size_t)r * SS + cc * 8, vb + r * VP + cc * 8);
    }
  };

  // Q A-fragments (16x32 f16 each), resident for the whole loop.
  // A layout: lane L holds row m=L%16; K = 8h+{0..7} then 16+8h+{0..7}.
  v16h Aq[4];
  {
    const _Float16* qbase = qh + ((size_t)b * SS + q0 + n) * DD;
    #pragma unroll
    for (int c = 0; c < 4; ++c)
      Aq[c] = ld16h2(qbase + c*32 + 8*h, qbase + c*32 + 16 + 8*h);
  }

  v8f O[8];
  #pragma unroll
  for (int g = 0; g < 8; ++g) O[g] = zero8();
  float mrow[8], lrow[8];
  #pragma unroll
  for (int j = 0; j < 8; ++j) { mrow[j] = -1e30f; lrow[j] = 0.0f; }

  const uint8_t* mbase = mask + ((size_t)b * SS + q0) * SS;
  _Float16* pw = &Ps[wave][0];

  stage(0, 0);                               // prime the pipeline

  for (int kt = 0; kt < NT; ++kt) {
    const int k0  = kt * KT;
    const int buf = kt & 1;

    if (kt + 1 < NT) {
      stage(kt + 1, buf ^ 1);                // prefetch next tile into other buffer
      WAIT_ASYNC(8);                         // in-order: current tile's 8 ops done
    } else {
      WAIT_ASYNC(0);
    }
    __syncthreads();                         // make staged LDS visible to all waves

    const _Float16* KsB = &Ks[buf][0];
    const _Float16* VsB = &Vs[buf][0];

    // S = Q * K^T (16x32): B-fragments in groups of 4 (keeps peak VGPRs < 256).
    // B layout: lane half h holds contraction K = 8h+{0..7} then 16+8h+{0..7}.
    v8f s0 = zero8(), s1 = zero8();
    #pragma unroll
    for (int cp = 0; cp < 2; ++cp) {
      v16h Bk[4];
      #pragma unroll
      for (int u = 0; u < 2; ++u) {
        const int c = 2*cp + u;
        Bk[2*u]   = ld16h2(KsB + n*KP      + c*32 + 8*h, KsB + n*KP      + c*32 + 16 + 8*h);
        Bk[2*u+1] = ld16h2(KsB + (16+n)*KP + c*32 + 8*h, KsB + (16+n)*KP + c*32 + 16 + 8*h);
      }
      #pragma unroll
      for (int u = 0; u < 2; ++u) {
        const int c = 2*cp + u;
        s0 = wmma_f16(Aq[c], Bk[2*u],   s0);
        s1 = wmma_f16(Aq[c], Bk[2*u+1], s1);
      }
    }

    // mask + online softmax.  C layout: VGPR j <-> row j + 8h, col = n (+16 frag1)
    float alpha[8];
    #pragma unroll
    for (int j = 0; j < 8; ++j) {
      const uint8_t* mr = mbase + ((size_t)(j + 8*h)) * SS + k0;
      float a0 = mr[n]      ? -1e30f : s0[j];
      float a1 = mr[16 + n] ? -1e30f : s1[j];
      float tmax = redmax16(fmaxf(a0, a1));
      float mnew = fmaxf(mrow[j], tmax);
      alpha[j]   = __expf(mrow[j] - mnew);
      mrow[j]    = mnew;
      float p0 = __expf(a0 - mnew);
      float p1 = __expf(a1 - mnew);
      s0[j] = p0; s1[j] = p1;
      lrow[j] = lrow[j] * alpha[j] + redsum16(p0 + p1);
    }

    // rescale accumulators
    #pragma unroll
    for (int g = 0; g < 8; ++g)
      #pragma unroll
      for (int j = 0; j < 8; ++j) O[g][j] *= alpha[j];

    // P: C-layout -> A-layout via per-wave LDS bounce (f32 -> f16)
    #pragma unroll
    for (int j = 0; j < 8; ++j) {
      pw[(j + 8*h)*PP + n]      = (_Float16)s0[j];
      pw[(j + 8*h)*PP + 16 + n] = (_Float16)s1[j];
    }
    v16h Ap = ld16h2(pw + n*PP + 8*h, pw + n*PP + 16 + 8*h);

    // O += P * V : B-fragments from V^T rows, in two groups of 4
    #pragma unroll
    for (int gp = 0; gp < 2; ++gp) {
      v16h Bv[4];
      #pragma unroll
      for (int u = 0; u < 4; ++u) {
        const int g = 4*gp + u;
        Bv[u] = ld16h2(VsB + (g*16 + n)*VP + 8*h, VsB + (g*16 + n)*VP + 16 + 8*h);
      }
      #pragma unroll
      for (int u = 0; u < 4; ++u)
        O[4*gp + u] = wmma_f16(Ap, Bv[u], O[4*gp + u]);
    }

    __syncthreads();                         // protect buffer before next prefetch reuses it
  }

  // epilogue: normalize and store fp32
  float* obase = out + ((size_t)b * SS + q0) * DD;
  #pragma unroll
  for (int j = 0; j < 8; ++j) {
    float inv = 1.0f / lrow[j];
    #pragma unroll
    for (int g = 0; g < 8; ++g)
      obase[(size_t)(j + 8*h)*DD + g*16 + n] = O[g][j] * inv;
  }
}

extern "C" void kernel_launch(void* const* d_in, const int* in_sizes, int n_in,
                              void* d_out, int out_size, void* d_ws, size_t ws_size,
                              hipStream_t stream) {
  const float*   q    = (const float*)d_in[0];
  const float*   k    = (const float*)d_in[1];
  const float*   v    = (const float*)d_in[2];
  const uint8_t* mask = (const uint8_t*)d_in[3];
  float*         out  = (float*)d_out;

  _Float16* qh = (_Float16*)d_ws;
  _Float16* kh = qh + (size_t)BB * SS * DD;
  _Float16* vt = kh + (size_t)BB * SS * DD;

  const int n  = BB * SS * DD;
  const int n4 = n / 4;
  const float qscale = 0.08838834764831845f;  // 1/sqrt(128)

  cvt_qk_kernel<<<(n4 + 255) / 256, 256, 0, stream>>>(
      (const float4*)q, (const float4*)k, (v4h*)qh, (v4h*)kh, n4, qscale);

  dim3 tb(32, 8), tg(SS / 32, DD / 32, BB);
  cvt_v_kernel<<<tg, tb, 0, stream>>>(v, vt);

  const int blocks = BB * (SS / QT) / WAVES;   // 512
  fa_kernel<<<blocks, TPB, 0, stream>>>(qh, kh, vt, mask, out);
}
